// EGNNConv_42511586296497
// MI455X (gfx1250) — matrix operand order repaired and checked by reading, not compiled, not run on hardware
//
#include <hip/hip_runtime.h>
#include <hip/hip_bf16.h>
#include <math.h>

typedef __attribute__((ext_vector_type(16))) _Float16 v16h;
typedef __attribute__((ext_vector_type(8)))  _Float16 v8h;
typedef __attribute__((ext_vector_type(8)))  float    v8f;

// LDS row strides (halves): padded so per-lane row step spreads across banks
#define S_WE1 304   // K=288 (257 padded) + 16
#define S_128 144   // K=128 + 16
#define S_WN1 272   // K=256 + 16
#define S_STG 136   // stage: K=128 + 8  (per-lane step = 4 banks -> conflict-free A reads)
#define TILES 4     // 16-edge/node tiles per wave per block

// ---------------------------------------------------------------- helpers
__device__ __forceinline__ float silu_f(float x) { return x / (1.0f + __expf(-x)); }

__device__ __forceinline__ v8f zero8f() {
    v8f r;
#pragma unroll
    for (int i = 0; i < 8; ++i) r[i] = 0.0f;
    return r;
}
__device__ __forceinline__ v8h zero8h() {
    v8h r;
#pragma unroll
    for (int i = 0; i < 8; ++i) r[i] = (_Float16)0.0f;
    return r;
}
__device__ __forceinline__ v16h cat16(v8h lo, v8h hi) {
    return __builtin_shufflevector(lo, hi, 0, 1, 2, 3, 4, 5, 6, 7, 8, 9, 10, 11, 12, 13, 14, 15);
}
__device__ __forceinline__ v8h cvt8(const float* __restrict__ p) {
    const float4 a = *reinterpret_cast<const float4*>(p);
    const float4 b = *reinterpret_cast<const float4*>(p + 4);
    v8h r;
    r[0] = (_Float16)a.x; r[1] = (_Float16)a.y; r[2] = (_Float16)a.z; r[3] = (_Float16)a.w;
    r[4] = (_Float16)b.x; r[5] = (_Float16)b.y; r[6] = (_Float16)b.z; r[7] = (_Float16)b.w;
    return r;
}
__device__ __forceinline__ v8f wmma_f16(v16h a, v16h b, v8f c) {
    return __builtin_amdgcn_wmma_f32_16x16x32_f16(false, a, false, b, (short)0, c, false, false);
}
// B fragment from LDS weight tile (f16, row n, padded row stride)
__device__ __forceinline__ v16h loadB_lds(const _Float16* sW, int stride, int nt, int kstep, int lane) {
    const int n  = nt * 16 + (lane & 15);
    const int ks = kstep * 32 + ((lane & 16) ? 8 : 0);
    const v8h* p = reinterpret_cast<const v8h*>(sW + n * stride + ks);
    return cat16(p[0], p[2]);
}
// A fragment from per-wave f16 LDS stage (16 rows x 128 K, row stride S_STG)
__device__ __forceinline__ v16h loadA_stage(const _Float16* st, int kstep, int lane) {
    const int m  = lane & 15;
    const int kb = kstep * 32 + ((lane & 16) ? 8 : 0);
    const v8h lo = *reinterpret_cast<const v8h*>(st + m * S_STG + kb);
    const v8h hi = *reinterpret_cast<const v8h*>(st + m * S_STG + kb + 16);
    return cat16(lo, hi);
}
// A K-run for edge layer 1: f = [nf[src](0..127) | nf[dst](128..255) | dist(256) | pad..287]
__device__ __forceinline__ v8h runA_edge(const float* __restrict__ nf, int s, int d, float dist, int kb) {
    if (kb < 128) return cvt8(nf + (size_t)s * 128 + kb);
    if (kb < 256) return cvt8(nf + (size_t)d * 128 + (kb - 128));
    v8h r = zero8h();
    if (kb == 256) r[0] = (_Float16)dist;
    return r;
}
// A K-run for node layer 1: [nf[node](0..127) | h_neigh[node](128..255)]
__device__ __forceinline__ v8h runA_node(const float* __restrict__ nf, const float* __restrict__ hacc,
                                         int node, int kb) {
    if (kb < 128) return cvt8(nf + (size_t)node * 128 + kb);
    return cvt8(hacc + (size_t)node * 128 + (kb - 128));
}
// cooperative 16B-chunk copy global->LDS
__device__ __forceinline__ void stage_copy(void* dst, const void* src, int bytes, int tid, int nthreads) {
    uint4* d = (uint4*)dst;
    const uint4* s = (const uint4*)src;
    const int cnt = bytes >> 4;
    for (int i = tid; i < cnt; i += nthreads) d[i] = s[i];
}

// ---------------------------------------------------------------- prep kernels
__global__ void zero_f32_kernel(float* __restrict__ p, long n) {
    long i = (long)blockIdx.x * blockDim.x + threadIdx.x;
    if (i < n) p[i] = 0.0f;
}
// W is f32 [K][128]; WT is f16 [128][Kstride], zero padded (matches LDS layout 1:1)
__global__ void transpose_w_kernel(const float* __restrict__ W, _Float16* __restrict__ WT, int K, int Kstride) {
    int idx = blockIdx.x * blockDim.x + threadIdx.x;
    int total = 128 * Kstride;
    if (idx >= total) return;
    int n = idx / Kstride, k = idx % Kstride;
    WT[idx] = (k < K) ? (_Float16)W[(size_t)k * 128 + n] : (_Float16)0.0f;
}

// ---------------------------------------------------------------- edge kernel
__global__ __launch_bounds__(256) void egnn_edge_kernel(
    const float* __restrict__ nf, const float* __restrict__ coord,
    const int* __restrict__ src, const int* __restrict__ dst,
    const _Float16* __restrict__ We1T, const float* __restrict__ be1,
    const _Float16* __restrict__ We2T, const float* __restrict__ be2,
    const _Float16* __restrict__ Wc1T, const float* __restrict__ bc1,
    const float* __restrict__ Wc2,
    float* __restrict__ hacc, float* __restrict__ xacc, float* __restrict__ cnt,
    int E)
{
    __shared__ __align__(16) _Float16 sWe1[128 * S_WE1];     // 77824 B
    __shared__ __align__(16) _Float16 sWe2[128 * S_128];     // 36864 B
    __shared__ __align__(16) _Float16 sWc1[128 * S_128];     // 36864 B
    __shared__ __align__(16) _Float16 stage[8][16 * S_STG];  // 34816 B
    __shared__ int   ssrc[8][16];
    __shared__ int   sdst[8][16];
    __shared__ float sdir[8][3][16];

    const int tid  = threadIdx.x;
    const int lane = tid & 31;
    const int wave = tid >> 5;
    const int half = (lane >> 4) & 1;
    const int ncol = lane & 15;

    // ---- stage all weights into LDS (shared by 8 waves x TILES tiles)
    stage_copy(sWe1, We1T, 128 * S_WE1 * 2, tid, 256);
    stage_copy(sWe2, We2T, 128 * S_128 * 2, tid, 256);
    stage_copy(sWc1, Wc1T, 128 * S_128 * 2, tid, 256);

    // hoist biases / Wc2 (per-lane column values)
    float vb1[8], vb2[8], vbc[8], vw2[8];
#pragma unroll
    for (int nt = 0; nt < 8; ++nt) {
        const int n = nt * 16 + ncol;
        vb1[nt] = be1[n]; vb2[nt] = be2[n]; vbc[nt] = bc1[n]; vw2[nt] = Wc2[n];
    }
    __syncthreads();

    _Float16* st = stage[wave];
    v8f acc[8];

    for (int t = 0; t < TILES; ++t) {
        const long ebase = ((long)blockIdx.x * (8 * TILES) + wave * TILES + t) * 16;
        if (ebase >= E) break;

        // ---- edge meta (lanes 0-15 compute, lanes 16-31 read via same-wave LDS)
        int s = 0, d = 0;
        float dist = 0.0f;
        if (lane < 16) {
            long e = ebase + lane;
            if (e < E) { s = src[e]; d = dst[e]; }
            float dx = coord[(size_t)s * 3 + 0] - coord[(size_t)d * 3 + 0];
            float dy = coord[(size_t)s * 3 + 1] - coord[(size_t)d * 3 + 1];
            float dz = coord[(size_t)s * 3 + 2] - coord[(size_t)d * 3 + 2];
            dist = sqrtf(dx * dx + dy * dy + dz * dz);
            float inv = 1.0f / (dist + 1e-7f);
            ssrc[wave][lane] = s;
            sdst[wave][lane] = d;
            sdir[wave][0][lane] = dx * inv;
            sdir[wave][1][lane] = dy * inv;
            sdir[wave][2][lane] = dz * inv;
        }
        if (lane >= 16) {
            s = ssrc[wave][lane - 16];
            d = sdst[wave][lane - 16];
        }

        // ---- layer 1: f(257, padded 288) @ We1
#pragma unroll
        for (int nt = 0; nt < 8; ++nt) acc[nt] = zero8f();
#pragma unroll
        for (int ks = 0; ks < 9; ++ks) {
            const int kb = ks * 32 + half * 8;
            v16h A = cat16(runA_edge(nf, s, d, dist, kb), runA_edge(nf, s, d, dist, kb + 16));
#pragma unroll
            for (int nt = 0; nt < 8; ++nt)
                acc[nt] = wmma_f16(A, loadB_lds(sWe1, S_WE1, nt, ks, lane), acc[nt]);
        }
#pragma unroll
        for (int nt = 0; nt < 8; ++nt) {
            const int n = nt * 16 + ncol;
#pragma unroll
            for (int r = 0; r < 8; ++r)
                st[(r + half * 8) * S_STG + n] = (_Float16)silu_f(acc[nt][r] + vb1[nt]);
        }

        // ---- layer 2: @ We2 -> msg_h (scatter + restage)
#pragma unroll
        for (int nt = 0; nt < 8; ++nt) acc[nt] = zero8f();
#pragma unroll
        for (int ks = 0; ks < 4; ++ks) {
            v16h A = loadA_stage(st, ks, lane);
#pragma unroll
            for (int nt = 0; nt < 8; ++nt)
                acc[nt] = wmma_f16(A, loadB_lds(sWe2, S_128, nt, ks, lane), acc[nt]);
        }
        int dl[8];
#pragma unroll
        for (int r = 0; r < 8; ++r) dl[r] = sdst[wave][r + half * 8];
#pragma unroll
        for (int nt = 0; nt < 8; ++nt) {
            const int n = nt * 16 + ncol;
#pragma unroll
            for (int r = 0; r < 8; ++r) {
                float v = silu_f(acc[nt][r] + vb2[nt]);
                st[(r + half * 8) * S_STG + n] = (_Float16)v;
                if (ebase + (r + half * 8) < E)
                    unsafeAtomicAdd(&hacc[(size_t)dl[r] * 128 + n], v);
            }
        }

        // ---- layer 3: msg_h @ Wc1 -> silu -> dot Wc2 -> coef
#pragma unroll
        for (int nt = 0; nt < 8; ++nt) acc[nt] = zero8f();
#pragma unroll
        for (int ks = 0; ks < 4; ++ks) {
            v16h A = loadA_stage(st, ks, lane);
#pragma unroll
            for (int nt = 0; nt < 8; ++nt)
                acc[nt] = wmma_f16(A, loadB_lds(sWc1, S_128, nt, ks, lane), acc[nt]);
        }
        float ps[8];
#pragma unroll
        for (int r = 0; r < 8; ++r) ps[r] = 0.0f;
#pragma unroll
        for (int nt = 0; nt < 8; ++nt) {
#pragma unroll
            for (int r = 0; r < 8; ++r) ps[r] += silu_f(acc[nt][r] + vbc[nt]) * vw2[nt];
        }
#pragma unroll
        for (int off = 8; off >= 1; off >>= 1) {
#pragma unroll
            for (int r = 0; r < 8; ++r) ps[r] += __shfl_xor(ps[r], off, 32);
        }
        if (ncol == 0) {
#pragma unroll
            for (int r = 0; r < 8; ++r) {
                const int m = r + half * 8;
                if (ebase + m < E) {
                    const int dd = sdst[wave][m];
                    const float coef = ps[r];
                    unsafeAtomicAdd(&xacc[(size_t)dd * 3 + 0], coef * sdir[wave][0][m]);
                    unsafeAtomicAdd(&xacc[(size_t)dd * 3 + 1], coef * sdir[wave][1][m]);
                    unsafeAtomicAdd(&xacc[(size_t)dd * 3 + 2], coef * sdir[wave][2][m]);
                    unsafeAtomicAdd(&cnt[dd], 1.0f);
                }
            }
        }
    }
}

// ---------------------------------------------------------------- node kernel
__global__ __launch_bounds__(256) void egnn_node_kernel(
    const float* __restrict__ nf, const float* __restrict__ coord,
    const _Float16* __restrict__ Wn1T, const float* __restrict__ bn1,
    const _Float16* __restrict__ Wn2T, const float* __restrict__ bn2,
    const float* __restrict__ hacc, const float* __restrict__ xacc, const float* __restrict__ cnt,
    float* __restrict__ out_h, float* __restrict__ out_x, int N)
{
    __shared__ __align__(16) _Float16 sWn1[128 * S_WN1];     // 69632 B
    __shared__ __align__(16) _Float16 sWn2[128 * S_128];     // 36864 B
    __shared__ __align__(16) _Float16 stage[8][16 * S_STG];  // 34816 B

    const int tid  = threadIdx.x;
    const int lane = tid & 31;
    const int wave = tid >> 5;
    const int half = (lane >> 4) & 1;
    const int ncol = lane & 15;

    stage_copy(sWn1, Wn1T, 128 * S_WN1 * 2, tid, 256);
    stage_copy(sWn2, Wn2T, 128 * S_128 * 2, tid, 256);

    float vb1[8], vb2[8];
#pragma unroll
    for (int nt = 0; nt < 8; ++nt) {
        const int n = nt * 16 + ncol;
        vb1[nt] = bn1[n]; vb2[nt] = bn2[n];
    }
    __syncthreads();

    _Float16* st = stage[wave];
    v8f acc[8];

    for (int t = 0; t < TILES; ++t) {
        const long nbase = ((long)blockIdx.x * (8 * TILES) + wave * TILES + t) * 16;
        if (nbase >= N) break;  // N multiple of 16: tiles are all-or-nothing
        const int node = (int)nbase + ncol;

        // ---- layer 1: [nf | h_neigh] (K=256) @ Wn1
#pragma unroll
        for (int nt = 0; nt < 8; ++nt) acc[nt] = zero8f();
#pragma unroll
        for (int ks = 0; ks < 8; ++ks) {
            const int kb = ks * 32 + half * 8;
            v16h A = cat16(runA_node(nf, hacc, node, kb), runA_node(nf, hacc, node, kb + 16));
#pragma unroll
            for (int nt = 0; nt < 8; ++nt)
                acc[nt] = wmma_f16(A, loadB_lds(sWn1, S_WN1, nt, ks, lane), acc[nt]);
        }
#pragma unroll
        for (int nt = 0; nt < 8; ++nt) {
#pragma unroll
            for (int r = 0; r < 8; ++r)
                st[(r + half * 8) * S_STG + nt * 16 + ncol] = (_Float16)silu_f(acc[nt][r] + vb1[nt]);
        }

        // ---- layer 2: @ Wn2 + bn2 (no activation) -> out_h
#pragma unroll
        for (int nt = 0; nt < 8; ++nt) acc[nt] = zero8f();
#pragma unroll
        for (int ks = 0; ks < 4; ++ks) {
            v16h A = loadA_stage(st, ks, lane);
#pragma unroll
            for (int nt = 0; nt < 8; ++nt)
                acc[nt] = wmma_f16(A, loadB_lds(sWn2, S_128, nt, ks, lane), acc[nt]);
        }
#pragma unroll
        for (int nt = 0; nt < 8; ++nt) {
            const int n = nt * 16 + ncol;
#pragma unroll
            for (int r = 0; r < 8; ++r) {
                const int m = r + half * 8;
                out_h[(size_t)(nbase + m) * 128 + n] = acc[nt][r] + vb2[nt];
            }
        }

        // ---- coord update: x + segment_mean(msg_x)
        if (lane < 16) {
            const int nd = (int)nbase + lane;
            const float inv = 1.0f / fmaxf(cnt[nd], 1.0f);
#pragma unroll
            for (int j = 0; j < 3; ++j)
                out_x[(size_t)nd * 3 + j] = coord[(size_t)nd * 3 + j] + xacc[(size_t)nd * 3 + j] * inv;
        }
    }
}

// ---------------------------------------------------------------- launch
extern "C" void kernel_launch(void* const* d_in, const int* in_sizes, int n_in,
                              void* d_out, int out_size, void* d_ws, size_t ws_size,
                              hipStream_t stream) {
    const float* nf    = (const float*)d_in[0];
    const float* coord = (const float*)d_in[1];
    const int*   src   = (const int*)d_in[2];
    const int*   dst   = (const int*)d_in[3];
    const float* We1 = (const float*)d_in[4],  *be1 = (const float*)d_in[5];
    const float* We2 = (const float*)d_in[6],  *be2 = (const float*)d_in[7];
    const float* Wn1 = (const float*)d_in[8],  *bn1 = (const float*)d_in[9];
    const float* Wn2 = (const float*)d_in[10], *bn2 = (const float*)d_in[11];
    const float* Wc1 = (const float*)d_in[12], *bc1 = (const float*)d_in[13];
    const float* Wc2 = (const float*)d_in[14];

    const int E = in_sizes[2];
    const int N = in_sizes[1] / 3;

    // workspace: padded f16 transposed weights (LDS-layout identical), then f32 accumulators
    _Float16* We1T = (_Float16*)d_ws;
    _Float16* We2T = We1T + 128 * S_WE1;
    _Float16* Wc1T = We2T + 128 * S_128;
    _Float16* Wn1T = Wc1T + 128 * S_128;
    _Float16* Wn2T = Wn1T + 128 * S_WN1;
    float* hacc = (float*)(Wn2T + 128 * S_128);
    float* xacc = hacc + (size_t)N * 128;
    float* cnt  = xacc + (size_t)N * 3;

    const long zn = (long)N * 128 + (long)N * 3 + N;
    zero_f32_kernel<<<(int)((zn + 255) / 256), 256, 0, stream>>>(hacc, zn);
    transpose_w_kernel<<<(128 * S_WE1 + 255) / 256, 256, 0, stream>>>(We1, We1T, 257, S_WE1);
    transpose_w_kernel<<<(128 * S_128 + 255) / 256, 256, 0, stream>>>(We2, We2T, 128, S_128);
    transpose_w_kernel<<<(128 * S_128 + 255) / 256, 256, 0, stream>>>(Wc1, Wc1T, 128, S_128);
    transpose_w_kernel<<<(128 * S_WN1 + 255) / 256, 256, 0, stream>>>(Wn1, Wn1T, 256, S_WN1);
    transpose_w_kernel<<<(128 * S_128 + 255) / 256, 256, 0, stream>>>(Wn2, Wn2T, 128, S_128);

    const int epb = 128 * TILES;  // edges per block
    const int eblocks = (E + epb - 1) / epb;
    egnn_edge_kernel<<<eblocks, 256, 0, stream>>>(nf, coord, src, dst,
                                                  We1T, be1, We2T, be2, Wc1T, bc1, Wc2,
                                                  hacc, xacc, cnt, E);

    float* out_h = (float*)d_out;
    float* out_x = out_h + (size_t)N * 128;
    const int npb = 128 * TILES;
    const int nblocks = (N + npb - 1) / npb;
    egnn_node_kernel<<<nblocks, 256, 0, stream>>>(nf, coord, Wn1T, bn1, Wn2T, bn2,
                                                  hacc, xacc, cnt, out_h, out_x, N);
}
